// HybridQLSTM_65481071408234
// MI455X (gfx1250) — matrix operand order repaired and checked by reading, not compiled, not run on hardware
//
#include <hip/hip_runtime.h>
#include <hip/hip_bf16.h>

#define NQ   4
#define BATCH 64
#define SEQ  512
#define EMB  128
#define TAG  10
// pre-activation buffer layout: pre[s][gate][b][wire] -> s*1024 + g*256 + b*4 + w
#define PRE_STRIDE_S 1024

typedef __attribute__((ext_vector_type(16))) _Float16 v16h;
typedef __attribute__((ext_vector_type(8)))  float    v8f;

// ---------------------------------------------------------------------------
// Phase 1: fused embedding-gather + 4 gate input projections via WMMA.
//   A = embeds rows (gathered from emb via x), shape [32768 x 128] (f16)
//   B = concat(Wf,Wi,Wu,Wo)[:128,:], shape [128 x 16]            (f16)
//   C = f32 accumulate; bias + theta folded into the store.
// One wave computes one 16x16 tile: 16 rows = 16 batches of one timestep s
// (B=64 -> 4 tiles per s), 16 cols = 4 gates x 4 wires. K fully unrolled:
// 4 x v_wmma_f32_16x16x32_f16 per tile, all gather loads issued up front.
// ---------------------------------------------------------------------------
__global__ __launch_bounds__(256) void qlstm_preact(
    const int*   __restrict__ x,   const float* __restrict__ emb,
    const float* __restrict__ Wf,  const float* __restrict__ Wi,
    const float* __restrict__ Wu,  const float* __restrict__ Wo,
    const float* __restrict__ bf,  const float* __restrict__ bi,
    const float* __restrict__ bu,  const float* __restrict__ bo,
    const float* __restrict__ thf, const float* __restrict__ thi,
    const float* __restrict__ thu, const float* __restrict__ tho,
    float* __restrict__ pre)
{
    const float* Ws[4] = {Wf, Wi, Wu, Wo};
    const float* Bs[4] = {bf, bi, bu, bo};
    const float* Ts[4] = {thf, thi, thu, tho};

    const int lane = threadIdx.x & 31;
    const int wid  = blockIdx.x * (blockDim.x >> 5) + (threadIdx.x >> 5); // tile id, 0..2047
    const int s    = wid >> 2;            // timestep
    const int b0   = (wid & 3) << 4;      // first batch row of tile
    const int mloc = lane & 15;           // M row / N col within tile for this lane
    const int hi   = lane >> 4;           // lane half (selects K sub-range)

    // A-side: gather one embedding row per M
    const int   brow = b0 + mloc;
    const int   idx  = x[brow * SEQ + s];
    const float* erow = emb + (long)idx * EMB;

    // B-side: this lane's output column -> (gate, wire)
    const int g = mloc >> 2, w = mloc & 3;
    const float* Wg = Ws[g];

    // bias + theta for this lane's column (overlap with tile compute)
    const float cb = Bs[g][w] + Ts[g][w];

    v8f acc = {};
    #pragma unroll
    for (int kb = 0; kb < EMB; kb += 32) {
        v16h av, bv;
        // A 16x32 f16, contiguous K order used here (same as B): lanes 0-15 hold K=0..15, lanes 16-31 hold K=16..31
        #pragma unroll
        for (int v = 0; v < 8; ++v) {
            int k = kb + 16 * hi + 2 * v;
            av[2 * v]     = (_Float16)erow[k];
            av[2 * v + 1] = (_Float16)erow[k + 1];
        }
        // B 32x16 f16, contiguous K order used here (same as A): lanes 0-15 hold K=0..15, lanes 16-31 hold K=16..31
        #pragma unroll
        for (int v = 0; v < 8; ++v) {
            int k = kb + 16 * hi + 2 * v;
            bv[2 * v]     = (_Float16)Wg[k * NQ + w];
            bv[2 * v + 1] = (_Float16)Wg[(k + 1) * NQ + w];
        }
        acc = __builtin_amdgcn_wmma_f32_16x16x32_f16(
            /*neg_a=*/false, av, /*neg_b=*/false, bv,
            /*c_mod=*/(short)0, acc, /*reuse_a=*/false, /*reuse_b=*/false);
    }

    // Store C (16x16 f32 layout: VGPR r -> M = r + 8*hi, N = mloc), fold bias+theta
    float* dst = pre + s * PRE_STRIDE_S + g * 256 + w;
    #pragma unroll
    for (int r = 0; r < 8; ++r) {
        int bb = b0 + r + 8 * hi;
        dst[bb * 4] = acc[r] + cb;
    }
}

// ---------------------------------------------------------------------------
// Phase 2: serial 512-step recurrence. One wave (32 lanes) = 8 batches x 4
// wires; 8 blocks cover B=64. All cross-wire mixing (4x4 recurrent matvec and
// the QLayer closed form) is done with in-wave shuffles: no LDS, no barriers.
// QLayer closed form (a = pre + h@Wrec, z_w = cos(a_w)):
//   Z0=z1 z2 z3, Z1=z0 z1, Z2=z0 z1 z2, Z3=z0 z1 z2 z3
// ---------------------------------------------------------------------------
__global__ __launch_bounds__(32) void qlstm_recur(
    const float* __restrict__ pre,
    const float* __restrict__ Wf, const float* __restrict__ Wi,
    const float* __restrict__ Wu, const float* __restrict__ Wo,
    float* __restrict__ hs)
{
    const float* Ws[4] = {Wf, Wi, Wu, Wo};
    const int lane  = threadIdx.x;          // 0..31
    const int n     = lane & 3;             // wire
    const int base  = lane & ~3;            // first lane of this batch's group
    const int bglob = blockIdx.x * 8 + (lane >> 2);

    // recurrent 4x4 weights for my wire n, all four gates
    float wrec[4][4];
    #pragma unroll
    for (int g = 0; g < 4; ++g)
        #pragma unroll
        for (int m = 0; m < 4; ++m)
            wrec[g][m] = Ws[g][(EMB + m) * NQ + n];

    float h = 0.f, c = 0.f;
    for (int s = 0; s < SEQ; ++s) {
        float zz[4];
        #pragma unroll
        for (int g = 0; g < 4; ++g) {
            float a = pre[s * PRE_STRIDE_S + g * 256 + bglob * 4 + n];
            #pragma unroll
            for (int m = 0; m < 4; ++m)
                a += __shfl(h, base + m, 32) * wrec[g][m];
            float cg = __cosf(a);
            float c0 = __shfl(cg, base + 0, 32);
            float c1 = __shfl(cg, base + 1, 32);
            float c2 = __shfl(cg, base + 2, 32);
            float c3 = __shfl(cg, base + 3, 32);
            float z1 = c0 * c1;
            float z2 = z1 * c2;
            float z3 = z2 * c3;
            float z0 = c1 * c2 * c3;
            zz[g] = (n == 0) ? z0 : (n == 1) ? z1 : (n == 2) ? z2 : z3;
        }
        float fg = 1.f / (1.f + __expf(-zz[0]));
        float ig = 1.f / (1.f + __expf(-zz[1]));
        float gg = tanhf(zz[2]);
        float og = 1.f / (1.f + __expf(-zz[3]));
        c = fg * c + ig * gg;
        h = og * tanhf(c);
        hs[s * (BATCH * NQ) + bglob * NQ + n] = h;   // coalesced per wave
    }
}

// ---------------------------------------------------------------------------
// Phase 3: tag projection + log_softmax. One thread per (b, s).
// ---------------------------------------------------------------------------
__global__ __launch_bounds__(256) void qlstm_tag(
    const float* __restrict__ hs, const float* __restrict__ Wt,
    const float* __restrict__ bt, float* __restrict__ out)
{
    const int r = blockIdx.x * 256 + threadIdx.x;   // 0 .. B*S-1
    const int b = r >> 9, s = r & (SEQ - 1);
    const float* hp = hs + s * (BATCH * NQ) + b * NQ;
    const float h0 = hp[0], h1 = hp[1], h2 = hp[2], h3 = hp[3];

    float l[TAG];
    float m = -3.0e38f;
    #pragma unroll
    for (int t = 0; t < TAG; ++t) {
        l[t] = bt[t] + h0 * Wt[t] + h1 * Wt[TAG + t]
                     + h2 * Wt[2 * TAG + t] + h3 * Wt[3 * TAG + t];
        m = fmaxf(m, l[t]);
    }
    float sum = 0.f;
    #pragma unroll
    for (int t = 0; t < TAG; ++t) sum += __expf(l[t] - m);
    const float lse = __logf(sum) + m;

    float* op = out + (long)(b * SEQ + s) * TAG;    // [B, S, TAG] row-major
    #pragma unroll
    for (int t = 0; t < TAG; ++t) op[t] = l[t] - lse;
}

// ---------------------------------------------------------------------------
extern "C" void kernel_launch(void* const* d_in, const int* in_sizes, int n_in,
                              void* d_out, int out_size, void* d_ws, size_t ws_size,
                              hipStream_t stream) {
    const int*   x   = (const int*)  d_in[0];
    const float* emb = (const float*)d_in[1];
    const float* Wf  = (const float*)d_in[2];
    const float* bf  = (const float*)d_in[3];
    const float* thf = (const float*)d_in[4];
    const float* Wi  = (const float*)d_in[5];
    const float* bi  = (const float*)d_in[6];
    const float* thi = (const float*)d_in[7];
    const float* Wu  = (const float*)d_in[8];
    const float* bu  = (const float*)d_in[9];
    const float* thu = (const float*)d_in[10];
    const float* Wo  = (const float*)d_in[11];
    const float* bo  = (const float*)d_in[12];
    const float* tho = (const float*)d_in[13];
    const float* Wt  = (const float*)d_in[14];
    const float* bt  = (const float*)d_in[15];
    float* out = (float*)d_out;

    float* pre = (float*)d_ws;                       // SEQ*1024 floats = 2 MB
    float* hs  = pre + SEQ * PRE_STRIDE_S;           // SEQ*256  floats = 0.5 MB

    // Phase 1: 2048 tiles, 8 waves per 256-thread block -> 256 blocks
    qlstm_preact<<<256, 256, 0, stream>>>(x, emb, Wf, Wi, Wu, Wo,
                                          bf, bi, bu, bo, thf, thi, thu, tho, pre);
    // Phase 2: 8 independent single-wave blocks (8 batches each)
    qlstm_recur<<<8, 32, 0, stream>>>(pre, Wf, Wi, Wu, Wo, hs);
    // Phase 3: 64*512 threads
    qlstm_tag<<<(BATCH * SEQ) / 256, 256, 0, stream>>>(hs, Wt, bt, out);
}
